// Encoder_2808908612228
// MI455X (gfx1250) — matrix operand order repaired and checked by reading, not compiled, not run on hardware
//
#include <hip/hip_runtime.h>

// ---------------------------------------------------------------------------
// Transformer encoder for MI455X (gfx1250, wave32, WMMA, async-to-LDS).
// bf16 operands are produced ONCE per tensor (epilogue/convert kernels), so
// GEMM/attention inner loops are pure async-copy + ds_load + v_wmma.
// ---------------------------------------------------------------------------

#define Bsz 16
#define Ssz 512
#define Dsz 512
#define Hh  8
#define KDs 64
#define VDs 64
#define HIDs 2048
#define Ls  4
#define OUTNs 128

typedef __attribute__((ext_vector_type(16))) __bf16 v16bf;
typedef __attribute__((ext_vector_type(8)))  float  v8f;
typedef __attribute__((ext_vector_type(4)))  int    v4i_;

union Frag16 { unsigned int u[8]; v16bf v; };

static __device__ __forceinline__ unsigned short f2bf(float x) {
  unsigned u = __float_as_uint(x);
  return (unsigned short)((u + 0x7FFFu + ((u >> 16) & 1u)) >> 16);
}

// A-fragment K index (16-bit A 16x32, ISA 7.12.2)
static __device__ __forceinline__ int a_kpair(int v, int half) {
  return (v < 4) ? (2 * v + 8 * half) : (16 + 2 * (v - 4) + 8 * half);
}
// B-fragment K index (same K order as A fragment)
static __device__ __forceinline__ int b_kpair(int v, int half) {
  return (v < 4) ? (2 * v + 8 * half) : (16 + 2 * (v - 4) + 8 * half);
}

static __device__ __forceinline__ v8f wmma_bf16(const Frag16& a, const Frag16& b, v8f c) {
  return __builtin_amdgcn_wmma_f32_16x16x32_bf16(false, a.v, false, b.v,
                                                 (short)0, c, false, false);
}

// ---- async global->LDS copy (ASYNCcnt path), with safe fallback ----------
#if __has_builtin(__builtin_amdgcn_global_load_async_to_lds_b128)
#define HAVE_ASYNC 1
#else
#define HAVE_ASYNC 0
#endif

static __device__ __forceinline__ void cp_g2l_b128(void* lds, const void* g) {
#if HAVE_ASYNC
  __builtin_amdgcn_global_load_async_to_lds_b128(
      (__attribute__((address_space(1))) v4i_*)g,
      (__attribute__((address_space(3))) v4i_*)lds, 0, 0);
#else
  *(uint4*)lds = *(const uint4*)g;
#endif
}
static __device__ __forceinline__ void wait_async() {
#if HAVE_ASYNC
#if __has_builtin(__builtin_amdgcn_s_wait_asynccnt)
  __builtin_amdgcn_s_wait_asynccnt(0);
#else
  asm volatile("s_wait_asynccnt 0x0" ::: "memory");
#endif
#endif
}

// ---------------------------------------------------------------------------
// 0) Weight convert+transpose: in f32 [K][N] -> out bf16 [N][K]
// ---------------------------------------------------------------------------
__global__ __launch_bounds__(256) void wcvt_t(const float* __restrict__ in,
                                              unsigned short* __restrict__ out,
                                              int K, int N) {
  __shared__ float tile[32][33];
  int bk = blockIdx.x * 32, bn = blockIdx.y * 32;
  int t = threadIdx.x;
  int r = t >> 3, c4 = (t & 7) * 4;
  const float* src = in + (long)(bk + r) * N + bn + c4;
  float4 f = *(const float4*)src;
  tile[r][c4] = f.x; tile[r][c4 + 1] = f.y; tile[r][c4 + 2] = f.z; tile[r][c4 + 3] = f.w;
  __syncthreads();
  unsigned short* dst = out + (long)(bn + r) * K + bk + c4;
  dst[0] = f2bf(tile[c4][r]);
  dst[1] = f2bf(tile[c4 + 1][r]);
  dst[2] = f2bf(tile[c4 + 2][r]);
  dst[3] = f2bf(tile[c4 + 3][r]);
}

// ---------------------------------------------------------------------------
// 1) Embedding gather + sinusoidal PE -> f32 x and bf16 xb
// ---------------------------------------------------------------------------
__global__ __launch_bounds__(256) void embed_pe(const int* __restrict__ tok,
                                                const float* __restrict__ emb,
                                                float* __restrict__ x,
                                                unsigned short* __restrict__ xb) {
  int bs = blockIdx.x;
  int s = bs & (Ssz - 1);
  int i = threadIdx.x;
  int tk = tok[bs];
  float dv = __expf(-(float)(2 * i) * 0.017988946f);   // ln(10000)/512
  float sv, cv;
  __sincosf((float)s * dv, &sv, &cv);
  const float* e = emb + (long)tk * Dsz;
  float v0 = e[2 * i] + sv, v1 = e[2 * i + 1] + cv;
  long o = (long)bs * Dsz + 2 * i;
  x[o] = v0;     x[o + 1] = v1;
  xb[o] = f2bf(v0); xb[o + 1] = f2bf(v1);
}

// ---------------------------------------------------------------------------
// 2) GEMM: act(A_bf16[M,K] @ Wt_bf16[N,K]^T + bias (+resid_f32)) -> Cf/Cb
//    128x64 block tile, 8 waves 4(M)x2(N), async staging into LDS.
// ---------------------------------------------------------------------------
__global__ __launch_bounds__(256) void gemm_wmma(const unsigned short* __restrict__ A,
                                                 const unsigned short* __restrict__ Wt,
                                                 const float* __restrict__ bias,
                                                 const float* __restrict__ resid,
                                                 float* __restrict__ Cf,
                                                 unsigned short* __restrict__ Cb,
                                                 int M, int N, int K, int relu) {
  __shared__ unsigned short ldsA[128][48];   // [m][k] bf16, 96B row (16B aligned)
  __shared__ unsigned short ldsW[64][48];    // [n][k] bf16
  int t = threadIdx.x;
  int lane = t & 31, w = t >> 5;
  int half = lane >> 4, lr = lane & 15;
  int wm = w >> 1, wn = w & 1;
  int bM = blockIdx.y * 128, bN = blockIdx.x * 64;

  v8f acc[2][2];
#pragma unroll
  for (int mt = 0; mt < 2; ++mt)
#pragma unroll
    for (int nt = 0; nt < 2; ++nt)
#pragma unroll
      for (int r = 0; r < 8; ++r) acc[mt][nt][r] = 0.0f;

  for (int k0 = 0; k0 < K; k0 += 32) {
    __syncthreads();
    // stage A tile 128x32 bf16: 512 x 16B segments, 2 per thread
#pragma unroll
    for (int i = 0; i < 2; ++i) {
      int seg = t * 2 + i, row = seg >> 2, j = seg & 3;
      cp_g2l_b128(&ldsA[row][j * 8], A + (long)(bM + row) * K + k0 + j * 8);
    }
    // stage W tile 64x32 bf16: 256 x 16B segments, 1 per thread
    {
      int row = t >> 2, j = t & 3;
      cp_g2l_b128(&ldsW[row][j * 8], Wt + (long)(bN + row) * K + k0 + j * 8);
    }
    if (k0 + 32 < K) {   // global_prefetch_b8 of next K tiles
      __builtin_prefetch(A + (long)(bM + (t >> 1)) * K + k0 + 32, 0, 1);
      __builtin_prefetch(Wt + (long)(bN + (t >> 2)) * K + k0 + 32, 0, 1);
    }
    wait_async();
    __syncthreads();

    Frag16 af[2], bf[2];
#pragma unroll
    for (int mt = 0; mt < 2; ++mt) {
      int row = wm * 32 + mt * 16 + lr;
#pragma unroll
      for (int v = 0; v < 8; ++v)
        af[mt].u[v] = *(const unsigned int*)&ldsA[row][a_kpair(v, half)];
    }
#pragma unroll
    for (int nt = 0; nt < 2; ++nt) {
      int col = wn * 32 + nt * 16 + lr;
#pragma unroll
      for (int v = 0; v < 8; ++v)
        bf[nt].u[v] = *(const unsigned int*)&ldsW[col][b_kpair(v, half)];
    }
#pragma unroll
    for (int mt = 0; mt < 2; ++mt)
#pragma unroll
      for (int nt = 0; nt < 2; ++nt)
        acc[mt][nt] = wmma_bf16(af[mt], bf[nt], acc[mt][nt]);
  }

#pragma unroll
  for (int mt = 0; mt < 2; ++mt)
#pragma unroll
    for (int nt = 0; nt < 2; ++nt)
#pragma unroll
      for (int r = 0; r < 8; ++r) {
        int row = bM + wm * 32 + mt * 16 + r + 8 * half;
        int col = bN + wn * 32 + nt * 16 + lr;
        float val = acc[mt][nt][r] + bias[col];
        if (resid) val += resid[(long)row * N + col];
        if (relu) val = fmaxf(val, 0.0f);
        if (Cf) Cf[(long)row * N + col] = val;
        if (Cb) Cb[(long)row * N + col] = f2bf(val);
      }
}

// ---------------------------------------------------------------------------
// 3) Flash attention (bf16 q/k/v in, bf16 ctx out)
// ---------------------------------------------------------------------------
__global__ __launch_bounds__(256) void attn_flash(const unsigned short* __restrict__ q,
                                                  const unsigned short* __restrict__ k,
                                                  const unsigned short* __restrict__ v,
                                                  const int* __restrict__ tok,
                                                  unsigned short* __restrict__ ctx) {
  __shared__ unsigned short ldsQ[128][72];      // [q][kd], 144B rows
  __shared__ unsigned short ldsK[64][72];       // [key][kd]
  __shared__ unsigned short ldsVT[64][66];      // [vd][key]
  __shared__ unsigned short ldsP[8][16][66];    // per-wave P tile
  __shared__ float ldsMask[64];
  int t = threadIdx.x, lane = t & 31, w = t >> 5;
  int half = lane >> 4, lr = lane & 15;
  int bid = blockIdx.x;
  int b = bid >> 5;
  int h = (bid >> 2) & 7;
  int qblk = (bid & 3) * 128;
  const float scale = 0.044194173824159216f;    // 1/sqrt(512)

  // stage Q 128x64 bf16: 1024 x 16B segments, 4 per thread (async)
#pragma unroll
  for (int i = 0; i < 4; ++i) {
    int seg = t * 4 + i, row = seg >> 3, j = seg & 7;
    cp_g2l_b128(&ldsQ[row][j * 8],
                q + (long)(b * Ssz + qblk + row) * (Hh * KDs) + h * KDs + j * 8);
  }
  wait_async();
  __syncthreads();

  Frag16 qf[2];
#pragma unroll
  for (int kc = 0; kc < 2; ++kc)
#pragma unroll
    for (int vv = 0; vv < 8; ++vv)
      qf[kc].u[vv] = *(const unsigned int*)&ldsQ[w * 16 + lr][kc * 32 + a_kpair(vv, half)];

  v8f accO[4];
#pragma unroll
  for (int nt = 0; nt < 4; ++nt)
#pragma unroll
    for (int r = 0; r < 8; ++r) accO[nt][r] = 0.0f;
  float Mrow[8], Lrow[8];
#pragma unroll
  for (int r = 0; r < 8; ++r) { Mrow[r] = -1e30f; Lrow[r] = 0.0f; }

  for (int jt = 0; jt < 8; ++jt) {
    int kb = jt * 64;
    __syncthreads();
    // K tile async: 512 segments, 2 per thread
#pragma unroll
    for (int i = 0; i < 2; ++i) {
      int seg = t * 2 + i, row = seg >> 3, j = seg & 7;
      cp_g2l_b128(&ldsK[row][j * 8],
                  k + (long)(b * Ssz + kb + row) * (Hh * KDs) + h * KDs + j * 8);
    }
    // V tile transposed (manual, pure bf16 moves)
    {
      int kr = t >> 2, cb = (t & 3) * 16;
      const unsigned short* sv = v + (long)(b * Ssz + kb + kr) * (Hh * VDs) + h * VDs + cb;
#pragma unroll
      for (int j = 0; j < 16; ++j) ldsVT[cb + j][kr] = sv[j];
    }
    if (t < 64) ldsMask[t] = (tok[b * Ssz + kb + t] == 0) ? -1e9f : 0.0f;
    wait_async();
    __syncthreads();

    v8f accS[4];
#pragma unroll
    for (int nt = 0; nt < 4; ++nt)
#pragma unroll
      for (int r = 0; r < 8; ++r) accS[nt][r] = 0.0f;
#pragma unroll
    for (int kc = 0; kc < 2; ++kc)
#pragma unroll
      for (int nt = 0; nt < 4; ++nt) {
        Frag16 kf;
#pragma unroll
        for (int vv = 0; vv < 8; ++vv)
          kf.u[vv] = *(const unsigned int*)&ldsK[nt * 16 + lr][kc * 32 + b_kpair(vv, half)];
        accS[nt] = wmma_bf16(qf[kc], kf, accS[nt]);
      }

    float mk[4];
#pragma unroll
    for (int nt = 0; nt < 4; ++nt) mk[nt] = ldsMask[nt * 16 + lr];

#pragma unroll
    for (int r = 0; r < 8; ++r) {
#pragma unroll
      for (int nt = 0; nt < 4; ++nt) accS[nt][r] = accS[nt][r] * scale + mk[nt];
      float mt = fmaxf(fmaxf(accS[0][r], accS[1][r]), fmaxf(accS[2][r], accS[3][r]));
      mt = fmaxf(mt, __shfl_xor(mt, 1));
      mt = fmaxf(mt, __shfl_xor(mt, 2));
      mt = fmaxf(mt, __shfl_xor(mt, 4));
      mt = fmaxf(mt, __shfl_xor(mt, 8));
      float mnew = fmaxf(Mrow[r], mt);
      float alpha = __expf(Mrow[r] - mnew);
      Mrow[r] = mnew;
      float ps = 0.0f;
#pragma unroll
      for (int nt = 0; nt < 4; ++nt) {
        float p = __expf(accS[nt][r] - mnew);
        accS[nt][r] = p;
        ps += p;
        accO[nt][r] *= alpha;
      }
      ps += __shfl_xor(ps, 1);
      ps += __shfl_xor(ps, 2);
      ps += __shfl_xor(ps, 4);
      ps += __shfl_xor(ps, 8);
      Lrow[r] = Lrow[r] * alpha + ps;
#pragma unroll
      for (int nt = 0; nt < 4; ++nt)
        ldsP[w][r + 8 * half][nt * 16 + lr] = f2bf(accS[nt][r]);
    }

#pragma unroll
    for (int kc = 0; kc < 2; ++kc) {
      Frag16 pf;
#pragma unroll
      for (int vv = 0; vv < 8; ++vv)
        pf.u[vv] = *(const unsigned int*)&ldsP[w][lr][kc * 32 + a_kpair(vv, half)];
#pragma unroll
      for (int nt = 0; nt < 4; ++nt) {
        Frag16 vf;
#pragma unroll
        for (int vv = 0; vv < 8; ++vv)
          vf.u[vv] = *(const unsigned int*)&ldsVT[nt * 16 + lr][kc * 32 + b_kpair(vv, half)];
        accO[nt] = wmma_bf16(pf, vf, accO[nt]);
      }
    }
  }

#pragma unroll
  for (int r = 0; r < 8; ++r) Lrow[r] = 1.0f / Lrow[r];
#pragma unroll
  for (int nt = 0; nt < 4; ++nt)
#pragma unroll
    for (int r = 0; r < 8; ++r) {
      int srow = qblk + w * 16 + r + 8 * half;
      ctx[(long)(b * Ssz + srow) * (Hh * VDs) + h * VDs + nt * 16 + lr] =
          f2bf(accO[nt][r] * Lrow[r]);
    }
}

// ---------------------------------------------------------------------------
// 4) LayerNorm over D=512 -> f32 out + bf16 outb
// ---------------------------------------------------------------------------
__global__ __launch_bounds__(256) void ln_kernel(const float* __restrict__ in,
                                                 const float* __restrict__ g,
                                                 const float* __restrict__ b,
                                                 float* __restrict__ out,
                                                 unsigned short* __restrict__ outb) {
  __shared__ float sd[256];
  int row = blockIdx.x;
  int t = threadIdx.x;
  const float* xr = in + (long)row * Dsz;
  float a0 = xr[t], a1 = xr[t + 256];
  sd[t] = a0 + a1;
  __syncthreads();
  for (int s = 128; s > 0; s >>= 1) { if (t < s) sd[t] += sd[t + s]; __syncthreads(); }
  float mean = sd[0] * (1.0f / Dsz);
  __syncthreads();
  float d0 = a0 - mean, d1 = a1 - mean;
  sd[t] = d0 * d0 + d1 * d1;
  __syncthreads();
  for (int s = 128; s > 0; s >>= 1) { if (t < s) sd[t] += sd[t + s]; __syncthreads(); }
  float rstd = rsqrtf(sd[0] * (1.0f / Dsz) + 1e-5f);
  float o0 = d0 * rstd * g[t] + b[t];
  float o1 = d1 * rstd * g[t + 256] + b[t + 256];
  out[(long)row * Dsz + t] = o0;
  out[(long)row * Dsz + t + 256] = o1;
  outb[(long)row * Dsz + t] = f2bf(o0);
  outb[(long)row * Dsz + t + 256] = f2bf(o1);
}

// ---------------------------------------------------------------------------
// 5) Head: split-K WMMA + ds_add_f32 + global_atomic_add_f32
// ---------------------------------------------------------------------------
__global__ void feat_init(const float* __restrict__ bfv, float* __restrict__ feats) {
  int i = blockIdx.x * 256 + threadIdx.x;
  if (i < Bsz * OUTNs) feats[i] = bfv[i & (OUTNs - 1)];
}

__global__ __launch_bounds__(256) void head1(const unsigned short* __restrict__ xb,
                                             const float* __restrict__ Wf,
                                             float* __restrict__ feats) {
  __shared__ float ldsC[16][16];
  int t = threadIdx.x, lane = t & 31, w = t >> 5;
  int half = lane >> 4, lr = lane & 15;
  int nt = blockIdx.x;
  long kbase = (long)blockIdx.y * 4096 + (long)w * 512;
  ldsC[t >> 4][t & 15] = 0.0f;
  __syncthreads();

  v8f acc;
#pragma unroll
  for (int r = 0; r < 8; ++r) acc[r] = 0.0f;

  for (int kc = 0; kc < 16; ++kc) {
    long k0 = kbase + kc * 32;
    Frag16 af, bfr;
#pragma unroll
    for (int vv = 0; vv < 8; ++vv) {
      long kk = k0 + a_kpair(vv, half);
      af.u[vv] = *(const unsigned int*)(xb + (long)lr * (Ssz * Dsz) + kk);
      long kb2 = k0 + b_kpair(vv, half);
      const float* pw = Wf + kb2 * OUTNs + nt * 16 + lr;
      bfr.u[vv] = (unsigned)f2bf(pw[0]) | ((unsigned)f2bf(pw[OUTNs]) << 16);
    }
    acc = wmma_bf16(af, bfr, acc);
  }
#pragma unroll
  for (int r = 0; r < 8; ++r) atomicAdd(&ldsC[r + 8 * half][lr], acc[r]);
  __syncthreads();
  atomicAdd(&feats[(t >> 4) * OUTNs + nt * 16 + (t & 15)], ldsC[t >> 4][t & 15]);
}

__global__ void head2(const float* __restrict__ feats, const float* __restrict__ Wout,
                      const float* __restrict__ bout, float* __restrict__ out) {
  int t = threadIdx.x;
  if (t < Bsz) {
    float s = bout[0];
    for (int j = 0; j < OUTNs; ++j) s += feats[t * OUTNs + j] * Wout[j];
    float y = 1.0f / (1.0f + __expf(-s));
    out[t] = y;
    out[Bsz + t] = rintf(y);
  }
  for (int i = t; i < Bsz * OUTNs; i += 256) out[2 * Bsz + i] = feats[i];
}

// ---------------------------------------------------------------------------
extern "C" void kernel_launch(void* const* d_in, const int* in_sizes, int n_in,
                              void* d_out, int out_size, void* d_ws, size_t ws_size,
                              hipStream_t stream) {
  const int*   tok  = (const int*)d_in[0];
  const float* emb  = (const float*)d_in[1];
  const float* Wq   = (const float*)d_in[2];
  const float* bq   = (const float*)d_in[3];
  const float* Wk   = (const float*)d_in[4];
  const float* bk   = (const float*)d_in[5];
  const float* Wv   = (const float*)d_in[6];
  const float* bv   = (const float*)d_in[7];
  const float* Wo   = (const float*)d_in[8];
  const float* bo   = (const float*)d_in[9];
  const float* ln1g = (const float*)d_in[10];
  const float* ln1b = (const float*)d_in[11];
  const float* W1   = (const float*)d_in[12];
  const float* b1   = (const float*)d_in[13];
  const float* W2   = (const float*)d_in[14];
  const float* b2   = (const float*)d_in[15];
  const float* ln2g = (const float*)d_in[16];
  const float* ln2b = (const float*)d_in[17];
  const float* Wf   = (const float*)d_in[18];
  const float* bfp  = (const float*)d_in[19];
  const float* Wout = (const float*)d_in[20];
  const float* bout = (const float*)d_in[21];

  const long NBS = (long)Bsz * Ssz;     // 8192
  const long XSZ = NBS * Dsz;           // 4,194,304
  float* x     = (float*)d_ws;
  float* t0    = x + XSZ;
  float* feats = t0 + XSZ;
  unsigned short* xb   = (unsigned short*)(feats + Bsz * OUTNs);
  unsigned short* qb16 = xb + XSZ;
  unsigned short* kb16 = qb16 + XSZ;
  unsigned short* vb16 = kb16 + XSZ;
  unsigned short* ctxb = vb16 + XSZ;
  unsigned short* hb16 = ctxb + XSZ;                    // 16,777,216
  unsigned short* wqT  = hb16 + NBS * HIDs;
  unsigned short* wkT  = wqT + (long)Ls * Dsz * Dsz;
  unsigned short* wvT  = wkT + (long)Ls * Dsz * Dsz;
  unsigned short* woT  = wvT + (long)Ls * Dsz * Dsz;
  unsigned short* w1T  = woT + (long)Ls * Dsz * Dsz;
  unsigned short* w2T  = w1T + (long)Ls * Dsz * HIDs;

  // one-shot weight convert+transpose (f32 [K][N] -> bf16 [N][K])
  for (int l = 0; l < Ls; ++l) {
    wcvt_t<<<dim3(16, 16), 256, 0, stream>>>(Wq + (long)l * Dsz * Dsz,  wqT + (long)l * Dsz * Dsz,  Dsz, Dsz);
    wcvt_t<<<dim3(16, 16), 256, 0, stream>>>(Wk + (long)l * Dsz * Dsz,  wkT + (long)l * Dsz * Dsz,  Dsz, Dsz);
    wcvt_t<<<dim3(16, 16), 256, 0, stream>>>(Wv + (long)l * Dsz * Dsz,  wvT + (long)l * Dsz * Dsz,  Dsz, Dsz);
    wcvt_t<<<dim3(16, 16), 256, 0, stream>>>(Wo + (long)l * Dsz * Dsz,  woT + (long)l * Dsz * Dsz,  Dsz, Dsz);
    wcvt_t<<<dim3(16, 64), 256, 0, stream>>>(W1 + (long)l * Dsz * HIDs, w1T + (long)l * Dsz * HIDs, Dsz, HIDs);
    wcvt_t<<<dim3(64, 16), 256, 0, stream>>>(W2 + (long)l * HIDs * Dsz, w2T + (long)l * HIDs * Dsz, HIDs, Dsz);
  }

  embed_pe<<<NBS, 256, 0, stream>>>(tok, emb, x, xb);

  for (int l = 0; l < Ls; ++l) {
    gemm_wmma<<<dim3(8, 64), 256, 0, stream>>>(xb, wqT + (long)l * Dsz * Dsz,
        bq + l * 512, nullptr, nullptr, qb16, 8192, 512, 512, 0);
    gemm_wmma<<<dim3(8, 64), 256, 0, stream>>>(xb, wkT + (long)l * Dsz * Dsz,
        bk + l * 512, nullptr, nullptr, kb16, 8192, 512, 512, 0);
    gemm_wmma<<<dim3(8, 64), 256, 0, stream>>>(xb, wvT + (long)l * Dsz * Dsz,
        bv + l * 512, nullptr, nullptr, vb16, 8192, 512, 512, 0);
    attn_flash<<<Bsz * Hh * (Ssz / 128), 256, 0, stream>>>(qb16, kb16, vb16, tok, ctxb);
    gemm_wmma<<<dim3(8, 64), 256, 0, stream>>>(ctxb, woT + (long)l * Dsz * Dsz,
        bo + l * 512, x, t0, nullptr, 8192, 512, 512, 0);
    ln_kernel<<<NBS, 256, 0, stream>>>(t0, ln1g + l * 512, ln1b + l * 512, x, xb);
    gemm_wmma<<<dim3(32, 64), 256, 0, stream>>>(xb, w1T + (long)l * Dsz * HIDs,
        b1 + l * HIDs, nullptr, nullptr, hb16, 8192, 2048, 512, 1);
    gemm_wmma<<<dim3(8, 64), 256, 0, stream>>>(hb16, w2T + (long)l * HIDs * Dsz,
        b2 + l * 512, x, t0, nullptr, 8192, 512, 2048, 0);
    ln_kernel<<<NBS, 256, 0, stream>>>(t0, ln2g + l * 512, ln2b + l * 512, x, xb);
  }

  feat_init<<<8, 256, 0, stream>>>(bfp, feats);
  head1<<<dim3(8, 64), 256, 0, stream>>>(xb, Wf, feats);
  head2<<<1, 256, 0, stream>>>(feats, Wout, bout, (float*)d_out);
}
